// NTM_83305185673878
// MI455X (gfx1250) — matrix operand order repaired and checked by reading, not compiled, not run on hardware
//
#include <hip/hip_runtime.h>
#include <math.h>

// ---------------- problem constants (from reference) ----------------
#define T_STEPS   32
#define BATCH     64
#define IN_SZ     32
#define H_SZ      256
#define N_SLOTS   2048
#define M_SZ      64
#define EPS_F     1e-8f
#define NTHREADS  256
#define NWAVES    8
#define RPT       (N_SLOTS / NTHREADS)   // rows per thread = 8

typedef __attribute__((ext_vector_type(16))) _Float16 v16h;
typedef __attribute__((ext_vector_type(8)))  float    v8f;

struct NTMParams {
  const float *x, *Wc, *bc;
  // [0]=write head, [1]=read head
  const float *Wk[2], *bk[2], *Wb[2], *bb[2], *Wg[2], *bg[2];
  const float *Ws[2], *bs[2], *Wgam[2], *bgam[2];
  const float *We, *be, *Wa, *ba, *Wo, *bo;
  const float *memory0, *ww0, *rw0, *read0;
  float *out;
  float *mem_ws;   // [BATCH][N_SLOTS][M_SZ] f32 scratch (L2-resident, 32 MB)
};

__device__ __forceinline__ float sigmoidf_(float x)  { return 1.f / (1.f + __expf(-x)); }
__device__ __forceinline__ float softplusf_(float x) { return x > 20.f ? x : log1pf(__expf(x)); }

__device__ __forceinline__ float dotH_(const float* ht, const float* w) {
  float a = 0.f;
  #pragma unroll 8
  for (int i = 0; i < H_SZ; ++i) a += ht[i] * w[i];
  return a;
}

// Deterministic block-wide tree reductions over 256 threads.
__device__ float block_sum(float v, float* buf) {
  const int tid = threadIdx.x;
  buf[tid] = v; __syncthreads();
  #pragma unroll
  for (int s = NTHREADS / 2; s > 0; s >>= 1) {
    if (tid < s) buf[tid] += buf[tid + s];
    __syncthreads();
  }
  float r = buf[0]; __syncthreads();
  return r;
}
__device__ float block_max(float v, float* buf) {
  const int tid = threadIdx.x;
  buf[tid] = v; __syncthreads();
  #pragma unroll
  for (int s = NTHREADS / 2; s > 0; s >>= 1) {
    if (tid < s) buf[tid] = fmaxf(buf[tid], buf[tid + s]);
    __syncthreads();
  }
  float r = buf[0]; __syncthreads();
  return r;
}

__global__ __launch_bounds__(NTHREADS)
void ntm_forward(NTMParams P)
{
  __shared__ __align__(16) float s_ww[N_SLOTS];     // write weights (persistent state)
  __shared__ __align__(16) float s_rw[N_SLOTS];     // read weights (persistent state)
  __shared__ __align__(16) float s_norm[N_SLOTS];   // ||memory row||
  __shared__ __align__(16) float s_numw[N_SLOTS];   // k_w . mem  (reused as wg scratch)
  __shared__ __align__(16) float s_numr[N_SLOTS];   // k_r . mem  (reused as wg scratch)
  __shared__ __align__(16) float s_ht[H_SZ];
  __shared__ __align__(16) float s_inp[IN_SZ + M_SZ];
  __shared__ __align__(16) float s_kw[M_SZ], s_kr[M_SZ];
  __shared__ __align__(16) float s_erase[M_SZ], s_add[M_SZ], s_read[M_SZ];
  __shared__             float s_scal[12];          // per head: beta,g,gamma,s0,s1,s2
  __shared__ __align__(16) float s_red[NTHREADS];
  __shared__ __align__(16) float s_wave[NWAVES][M_SZ];

  const int tid  = threadIdx.x;
  const int lane = tid & 31;
  const int wv   = tid >> 5;      // wave id 0..7
  const int col  = lane & 15;     // WMMA N-column / A-row index
  const int hi   = lane >> 4;     // lane half (K-split per ISA layout)
  const int b    = blockIdx.x;

  float* memB = P.mem_ws + (size_t)b * N_SLOTS * M_SZ;

  // ---------------- init: memory0 -> scratch (+norms), softmax(ww0/rw0), read0
  for (int r = tid; r < N_SLOTS; r += NTHREADS) {
    const float4* src = (const float4*)(P.memory0 + (size_t)r * M_SZ);
    float4*       dst = (float4*)(memB + (size_t)r * M_SZ);
    float nrm = 0.f;
    #pragma unroll
    for (int q = 0; q < M_SZ / 4; ++q) {
      float4 v = src[q];
      dst[q] = v;
      nrm += v.x * v.x + v.y * v.y + v.z * v.z + v.w * v.w;
    }
    s_norm[r] = sqrtf(nrm);
  }
  for (int h = 0; h < 2; ++h) {
    const float* w0  = h ? P.rw0 : P.ww0;
    float*       dst = h ? s_rw  : s_ww;
    float c[RPT]; float lmax = -3.4e38f;
    #pragma unroll
    for (int k = 0; k < RPT; ++k) { c[k] = w0[tid + k * NTHREADS]; lmax = fmaxf(lmax, c[k]); }
    float vmax = block_max(lmax, s_red);
    float ls = 0.f;
    #pragma unroll
    for (int k = 0; k < RPT; ++k) { c[k] = __expf(c[k] - vmax); ls += c[k]; }
    float inv = 1.f / block_sum(ls, s_red);
    #pragma unroll
    for (int k = 0; k < RPT; ++k) dst[tid + k * NTHREADS] = c[k] * inv;
  }
  if (tid < M_SZ) s_read[tid] = P.read0[tid];
  __syncthreads();

  // ---------------- timestep recurrence ----------------
  for (int t = 0; t < T_STEPS; ++t) {
    // A: build controller input [x_t | read]
    if (tid < IN_SZ)                s_inp[tid] = P.x[((size_t)t * BATCH + b) * IN_SZ + tid];
    else if (tid < IN_SZ + M_SZ)    s_inp[tid] = s_read[tid - IN_SZ];
    __syncthreads();

    // B: ht = tanh(inp @ Wc + bc), one output per thread
    {
      float acc = P.bc[tid];
      #pragma unroll 4
      for (int i = 0; i < IN_SZ + M_SZ; ++i) acc += s_inp[i] * P.Wc[i * H_SZ + tid];
      s_ht[tid] = tanhf(acc);
    }
    __syncthreads();

    // C: head projections (k_w, k_r, erase, add) — 64 threads per output vector
    {
      const int grp = tid >> 6, l = tid & 63;
      const float *Wm, *bv; float* dst; int sig;
      if      (grp == 0) { Wm = P.Wk[0]; bv = P.bk[0]; dst = s_kw;    sig = 0; }
      else if (grp == 1) { Wm = P.Wk[1]; bv = P.bk[1]; dst = s_kr;    sig = 0; }
      else if (grp == 2) { Wm = P.We;    bv = P.be;    dst = s_erase; sig = 1; }
      else               { Wm = P.Wa;    bv = P.ba;    dst = s_add;   sig = 1; }
      float acc = bv[l];
      #pragma unroll 8
      for (int hh = 0; hh < H_SZ; ++hh) acc += s_ht[hh] * Wm[hh * M_SZ + l];
      dst[l] = sig ? sigmoidf_(acc) : tanhf(acc);
    }
    // scalar gates per head (threads 0..15, sequential extra work)
    if (tid < 16) {
      const int h = tid >> 3, which = tid & 7;
      if (which == 0)      s_scal[h * 6 + 0] = softplusf_(dotH_(s_ht, P.Wb[h])   + P.bb[h][0]);
      else if (which == 1) s_scal[h * 6 + 1] = sigmoidf_(dotH_(s_ht, P.Wg[h])    + P.bg[h][0]);
      else if (which == 2) s_scal[h * 6 + 2] = 1.f + softplusf_(dotH_(s_ht, P.Wgam[h]) + P.bgam[h][0]);
      else if (which == 3) {
        float l0 = P.bs[h][0], l1 = P.bs[h][1], l2 = P.bs[h][2];
        for (int hh = 0; hh < H_SZ; ++hh) {
          float v = s_ht[hh]; const float* w = P.Ws[h] + hh * 3;
          l0 += v * w[0]; l1 += v * w[1]; l2 += v * w[2];
        }
        float m = fmaxf(l0, fmaxf(l1, l2));
        float e0 = __expf(l0 - m), e1 = __expf(l1 - m), e2 = __expf(l2 - m);
        float is = 1.f / (e0 + e1 + e2);
        s_scal[h * 6 + 3] = e0 * is; s_scal[h * 6 + 4] = e1 * is; s_scal[h * 6 + 5] = e2 * is;
      }
    }
    __syncthreads();
    const float knw = sqrtf(block_sum(tid < M_SZ ? s_kw[tid] * s_kw[tid] : 0.f, s_red));
    const float knr = sqrtf(block_sum(tid < M_SZ ? s_kr[tid] * s_kr[tid] : 0.f, s_red));

    // D: content similarities via WMMA: num[2048,2] = memory[2048,64] @ [k_w | k_r]
    {
      // B fragment (32x16 f16): N = col, K = 8*hi + (e&7) + 16*(e>>3) (matches A layout); cols 0/1 = k_w/k_r
      v16h bf0, bf1;
      #pragma unroll
      for (int e = 0; e < 16; ++e) {
        const int k0 = hi * 8 + (e & 7) + (e >> 3) * 16;
        float v0 = (col == 0) ? s_kw[k0]      : (col == 1) ? s_kr[k0]      : 0.f;
        float v1 = (col == 0) ? s_kw[32 + k0] : (col == 1) ? s_kr[32 + k0] : 0.f;
        bf0[e] = (_Float16)v0;
        bf1[e] = (_Float16)v1;
      }
      const int kb = hi * 8;  // A layout: lane half selects K runs {kb..kb+7, kb+16..kb+23}
      for (int tile = wv; tile < N_SLOTS / 16; tile += NWAVES) {
        const float* rp = memB + (size_t)(tile * 16 + col) * M_SZ;
        if (tile + NWAVES < N_SLOTS / 16)
          __builtin_prefetch(memB + (size_t)((tile + NWAVES) * 16 + col) * M_SZ, 0, 1);
        v16h a0, a1;
        #pragma unroll
        for (int e = 0; e < 8; ++e) {
          a0[e]     = (_Float16)rp[kb + e];
          a0[e + 8] = (_Float16)rp[kb + 16 + e];
          a1[e]     = (_Float16)rp[32 + kb + e];
          a1[e + 8] = (_Float16)rp[32 + kb + 16 + e];
        }
        v8f d = {};
        d = __builtin_amdgcn_wmma_f32_16x16x32_f16(false, a0, false, bf0, (short)0, d, false, false);
        d = __builtin_amdgcn_wmma_f32_16x16x32_f16(false, a1, false, bf1, (short)0, d, false, false);
        const int rbase = tile * 16 + hi * 8;  // D layout: vgpr j -> row j + 8*hi, col = lane&15
        if (col == 0) {
          #pragma unroll
          for (int j = 0; j < 8; ++j) s_numw[rbase + j] = d[j];
        } else if (col == 1) {
          #pragma unroll
          for (int j = 0; j < 8; ++j) s_numr[rbase + j] = d[j];
        }
      }
    }
    __syncthreads();

    // E: addressing (content softmax -> interpolate -> shift -> sharpen) per head
    for (int h = 0; h < 2; ++h) {
      const float* num   = h ? s_numr : s_numw;
      float*       wgbuf = h ? s_numr : s_numw;   // reuse after num consumed
      float*       wdst  = h ? s_rw   : s_ww;
      const float beta  = s_scal[h * 6 + 0];
      const float g     = s_scal[h * 6 + 1];
      const float gamma = s_scal[h * 6 + 2];
      const float sh0   = s_scal[h * 6 + 3];
      const float sh1   = s_scal[h * 6 + 4];
      const float sh2   = s_scal[h * 6 + 5];
      const float kn    = h ? knr : knw;

      float c[RPT]; float lmax = -3.4e38f;
      #pragma unroll
      for (int k = 0; k < RPT; ++k) {
        const int n = tid + k * NTHREADS;
        c[k] = beta * num[n] / (kn * s_norm[n] + EPS_F);
        lmax = fmaxf(lmax, c[k]);
      }
      float vmax = block_max(lmax, s_red);
      float ls = 0.f;
      #pragma unroll
      for (int k = 0; k < RPT; ++k) { c[k] = __expf(c[k] - vmax); ls += c[k]; }
      float inv = 1.f / block_sum(ls, s_red);
      #pragma unroll
      for (int k = 0; k < RPT; ++k) {
        const int n = tid + k * NTHREADS;
        c[k] = g * (c[k] * inv) + (1.f - g) * wdst[n];   // interpolation wg
      }
      __syncthreads();
      #pragma unroll
      for (int k = 0; k < RPT; ++k) wgbuf[tid + k * NTHREADS] = c[k];
      __syncthreads();
      float p[RPT]; float ps = 0.f;
      #pragma unroll
      for (int k = 0; k < RPT; ++k) {
        const int n = tid + k * NTHREADS;
        float wt = sh0 * wgbuf[(n + 1) & (N_SLOTS - 1)]
                 + sh1 * wgbuf[n]
                 + sh2 * wgbuf[(n - 1) & (N_SLOTS - 1)];
        p[k] = __expf(gamma * __logf(wt + EPS_F));        // (wt+eps)^gamma
        ps += p[k];
      }
      float ipn = 1.f / block_sum(ps, s_red);
      #pragma unroll
      for (int k = 0; k < RPT; ++k) wdst[tid + k * NTHREADS] = p[k] * ipn;
      __syncthreads();
    }

    // F: memory erase/add update + new row norms + read = rw @ memory_new
    {
      float rac[M_SZ];
      #pragma unroll
      for (int m = 0; m < M_SZ; ++m) rac[m] = 0.f;
      for (int r = tid; r < N_SLOTS; r += NTHREADS) {
        const float wwn = s_ww[r], rwn = s_rw[r];
        float4*       rp = (float4*)(memB + (size_t)r * M_SZ);
        const float4* ep = (const float4*)s_erase;
        const float4* ap = (const float4*)s_add;
        float nrm = 0.f;
        #pragma unroll
        for (int q = 0; q < M_SZ / 4; ++q) {
          float4 v = rp[q], e = ep[q], a = ap[q];
          v.x = v.x * (1.f - wwn * e.x) + wwn * a.x;
          v.y = v.y * (1.f - wwn * e.y) + wwn * a.y;
          v.z = v.z * (1.f - wwn * e.z) + wwn * a.z;
          v.w = v.w * (1.f - wwn * e.w) + wwn * a.w;
          rp[q] = v;
          nrm += v.x * v.x + v.y * v.y + v.z * v.z + v.w * v.w;
          rac[q * 4 + 0] += rwn * v.x;
          rac[q * 4 + 1] += rwn * v.y;
          rac[q * 4 + 2] += rwn * v.z;
          rac[q * 4 + 3] += rwn * v.w;
        }
        s_norm[r] = sqrtf(nrm);
      }
      // deterministic reduction: wave32 butterfly, then serial 8-way cross-wave sum
      #pragma unroll
      for (int m = 0; m < M_SZ; ++m) {
        float v = rac[m];
        v += __shfl_xor(v, 16); v += __shfl_xor(v, 8); v += __shfl_xor(v, 4);
        v += __shfl_xor(v, 2);  v += __shfl_xor(v, 1);
        if (lane == 0) s_wave[wv][m] = v;
      }
      __syncthreads();
      if (tid < M_SZ) {
        float sum = 0.f;
        #pragma unroll
        for (int w = 0; w < NWAVES; ++w) sum += s_wave[w][tid];
        s_read[tid] = sum;
      }
      __syncthreads();
    }

    // G: out_t = sigmoid(read @ Wo + bo)
    if (tid < IN_SZ) {
      float acc = P.bo[tid];
      #pragma unroll 4
      for (int m = 0; m < M_SZ; ++m) acc += s_read[m] * P.Wo[m * IN_SZ + tid];
      P.out[((size_t)t * BATCH + b) * IN_SZ + tid] = sigmoidf_(acc);
    }
    __syncthreads();
  }
}

extern "C" void kernel_launch(void* const* d_in, const int* in_sizes, int n_in,
                              void* d_out, int out_size, void* d_ws, size_t ws_size,
                              hipStream_t stream) {
  (void)in_sizes; (void)n_in; (void)out_size; (void)ws_size;
  const float* const* in = (const float* const*)d_in;

  NTMParams P;
  // setup_inputs() insertion order, head dicts flattened in _head_params order.
  P.x  = in[0];  P.Wc = in[1];  P.bc = in[2];
  for (int h = 0; h < 2; ++h) {
    const int o = 3 + h * 10;   // 3..12 write head, 13..22 read head
    P.Wk[h]   = in[o + 0]; P.bk[h]   = in[o + 1];
    P.Wb[h]   = in[o + 2]; P.bb[h]   = in[o + 3];
    P.Wg[h]   = in[o + 4]; P.bg[h]   = in[o + 5];
    P.Ws[h]   = in[o + 6]; P.bs[h]   = in[o + 7];
    P.Wgam[h] = in[o + 8]; P.bgam[h] = in[o + 9];
  }
  P.We = in[23]; P.be = in[24];
  P.Wa = in[25]; P.ba = in[26];
  P.Wo = in[27]; P.bo = in[28];
  P.memory0 = in[29]; P.ww0 = in[30]; P.rw0 = in[31]; P.read0 = in[32];
  P.out    = (float*)d_out;
  P.mem_ws = (float*)d_ws;   // needs BATCH*N*M*4 = 32 MB (L2-resident)

  ntm_forward<<<dim3(BATCH), dim3(NTHREADS), 0, stream>>>(P);
}
